// ATT_learner_4080218931471
// MI455X (gfx1250) — matrix operand order repaired and checked by reading, not compiled, not run on hardware
//
#include <hip/hip_runtime.h>
#include <hip/hip_fp16.h>
#include <math.h>

#define NN    16384
#define ISIZE 512
#define KNN   31          // K+1 neighbors kept per node
#define NTILE 512         // block column tile (8 waves x 64)
#define LDW   520         // padded LDS row stride (floats) to dodge bank conflicts

typedef __attribute__((ext_vector_type(16))) _Float16 v16h;
typedef __attribute__((ext_vector_type(8)))  _Float16 v8h;
typedef __attribute__((ext_vector_type(8)))  float    v8f;

// ---------------------------------------------------------------------------
// Phase A: emb = relu(f*w0)*w1 ; L2-normalize rows ; store as f16 (row-major)
// One block per row, 256 threads, 2 elements per thread.
// ---------------------------------------------------------------------------
__global__ __launch_bounds__(256) void embed_normalize_kernel(
    const float* __restrict__ F, const float* __restrict__ w0,
    const float* __restrict__ w1, _Float16* __restrict__ E) {
  __shared__ float red[8];
  const int row = blockIdx.x;
  const int t   = threadIdx.x;
  const float* fr = F + (size_t)row * ISIZE;

  float a = fr[t]       * w0[t];
  float b = fr[t + 256] * w0[t + 256];
  a = fmaxf(a, 0.f) * w1[t];
  b = fmaxf(b, 0.f) * w1[t + 256];

  float s = a * a + b * b;
  #pragma unroll
  for (int off = 16; off > 0; off >>= 1) s += __shfl_xor(s, off, 32);
  if ((t & 31) == 0) red[t >> 5] = s;
  __syncthreads();
  if (t == 0) {
    float tot = 0.f;
    #pragma unroll
    for (int i = 0; i < 8; ++i) tot += red[i];
    red[0] = 1.0f / fmaxf(sqrtf(tot), 1e-12f);
  }
  __syncthreads();
  const float sc = red[0];
  _Float16* er = E + (size_t)row * ISIZE;
  er[t]       = (_Float16)(a * sc);
  er[t + 256] = (_Float16)(b * sc);
}

// ---------------------------------------------------------------------------
// Phase B: fused sim = E*E^T (f16 WMMA, f32 accum) + streaming top-31 per row.
// Block = 256 threads = 8 waves. Each wave computes a 32(M)x64(N) sub-tile
// (2 A-fragments x 4 B-fragments -> 8 accumulators), so every B fragment
// feeds two WMMAs. Block tile = 32 x 512; 32 j-steps sweep all columns.
// Top-31 lists live in LDS; insertion is ballot-filtered vs row threshold.
// ---------------------------------------------------------------------------
__global__ __launch_bounds__(256) void simtopk_kernel(
    const _Float16* __restrict__ E,
    int* __restrict__ rowsOut, int* __restrict__ colsOut,
    float* __restrict__ valsOut) {
  __shared__ float s_sim[32 * LDW];
  __shared__ float s_val[32 * KNN];
  __shared__ int   s_col[32 * KNN];

  const int t    = threadIdx.x;
  const int lane = t & 31;
  const int w    = t >> 5;      // wave id 0..7 -> owns columns [64w, 64w+64)
  const int hs   = lane >> 4;   // half-wave select
  const int l16  = lane & 15;
  const int i0   = blockIdx.x * 32;

  for (int q = t; q < 32 * KNN; q += 256) {
    s_val[q] = -3.4e38f;
    s_col[q] = 0;
  }
  __syncthreads();

  // A fragment sources: rows (i0 + l16) and (i0 + 16 + l16) of E.
  const _Float16* a0Base = E + (size_t)(i0 +      l16) * ISIZE;
  const _Float16* a1Base = E + (size_t)(i0 + 16 + l16) * ISIZE;

  for (int j0 = 0; j0 < NN; j0 += NTILE) {
    v8f c00 = {}, c01 = {}, c02 = {}, c03 = {};
    v8f c10 = {}, c11 = {}, c12 = {}, c13 = {};
    // B fragment sources: rows of E serve directly as B columns (C = E*E^T).
    const _Float16* b0r = E + (size_t)(j0 + 64 * w +  0 + l16) * ISIZE + 16 * hs;
    const _Float16* b1r = E + (size_t)(j0 + 64 * w + 16 + l16) * ISIZE + 16 * hs;
    const _Float16* b2r = E + (size_t)(j0 + 64 * w + 32 + l16) * ISIZE + 16 * hs;
    const _Float16* b3r = E + (size_t)(j0 + 64 * w + 48 + l16) * ISIZE + 16 * hs;

    #pragma unroll 2
    for (int k = 0; k < ISIZE; k += 32) {
      v16h a0, a1;
      {
        a0 = *(const v16h*)(a0Base + k + 16 * hs);
        a1 = *(const v16h*)(a1Base + k + 16 * hs);
      }
      v16h b;
      b   = *(const v16h*)(b0r + k);
      c00 = __builtin_amdgcn_wmma_f32_16x16x32_f16(false, a0, false, b, (short)0, c00, false, false);
      c10 = __builtin_amdgcn_wmma_f32_16x16x32_f16(false, a1, false, b, (short)0, c10, false, false);
      b   = *(const v16h*)(b1r + k);
      c01 = __builtin_amdgcn_wmma_f32_16x16x32_f16(false, a0, false, b, (short)0, c01, false, false);
      c11 = __builtin_amdgcn_wmma_f32_16x16x32_f16(false, a1, false, b, (short)0, c11, false, false);
      b   = *(const v16h*)(b2r + k);
      c02 = __builtin_amdgcn_wmma_f32_16x16x32_f16(false, a0, false, b, (short)0, c02, false, false);
      c12 = __builtin_amdgcn_wmma_f32_16x16x32_f16(false, a1, false, b, (short)0, c12, false, false);
      b   = *(const v16h*)(b3r + k);
      c03 = __builtin_amdgcn_wmma_f32_16x16x32_f16(false, a0, false, b, (short)0, c03, false, false);
      c13 = __builtin_amdgcn_wmma_f32_16x16x32_f16(false, a1, false, b, (short)0, c13, false, false);
    }

    // Spill 16x16 C tiles to LDS: lane l16 = column, VGPR r -> row 8*hs + r.
    {
      const int rb0   = 8 * hs;        // rows of A-tile 0
      const int rb1   = 16 + 8 * hs;   // rows of A-tile 1
      const int cbase = 64 * w + l16;
      #pragma unroll
      for (int r = 0; r < 8; ++r) {
        s_sim[(rb0 + r) * LDW + cbase +  0] = c00[r];
        s_sim[(rb0 + r) * LDW + cbase + 16] = c01[r];
        s_sim[(rb0 + r) * LDW + cbase + 32] = c02[r];
        s_sim[(rb0 + r) * LDW + cbase + 48] = c03[r];
        s_sim[(rb1 + r) * LDW + cbase +  0] = c10[r];
        s_sim[(rb1 + r) * LDW + cbase + 16] = c11[r];
        s_sim[(rb1 + r) * LDW + cbase + 32] = c12[r];
        s_sim[(rb1 + r) * LDW + cbase + 48] = c13[r];
      }
    }
    __syncthreads();

    // Top-k scan: wave w owns local rows 4w..4w+3; ballot-filter candidates,
    // lane 0 does the (rare) sorted insertions.
    #pragma unroll
    for (int rr = 0; rr < 4; ++rr) {
      const int r = 4 * w + rr;
      for (int ch = 0; ch < NTILE / 32; ++ch) {
        const int cLoc = ch * 32 + lane;
        const float v   = s_sim[r * LDW + cLoc];
        const float thr = s_val[r * KNN + 30];
        unsigned long long m = __ballot(v > thr);
        while (m) {
          const int src  = __ffsll((long long)m) - 1;
          const float vv = __shfl(v, src, 32);
          const int   cc = __shfl(cLoc, src, 32);
          if (lane == 0) {
            if (vv > s_val[r * KNN + 30]) {   // re-check vs fresh threshold
              int p = 30;
              while (p > 0 && s_val[r * KNN + p - 1] < vv) {
                s_val[r * KNN + p] = s_val[r * KNN + p - 1];
                s_col[r * KNN + p] = s_col[r * KNN + p - 1];
                --p;
              }
              s_val[r * KNN + p] = vv;
              s_col[r * KNN + p] = j0 + cc;
            }
          }
          m &= m - 1;
        }
      }
    }
    __syncthreads();
  }

  // Emit (rows, cols, relu(values)); lists already sorted descending.
  for (int q = t; q < 32 * KNN; q += 256) {
    const int r  = q / KNN;
    const int p  = q % KNN;
    const int gi = i0 + r;
    const int o  = gi * KNN + p;
    rowsOut[o] = gi;
    colsOut[o] = s_col[q];
    valsOut[o] = fmaxf(s_val[q], 0.f);
  }
}

// ---------------------------------------------------------------------------
extern "C" void kernel_launch(void* const* d_in, const int* in_sizes, int n_in,
                              void* d_out, int out_size, void* d_ws, size_t ws_size,
                              hipStream_t stream) {
  const float* F  = (const float*)d_in[0];   // [N, 512]
  const float* w0 = (const float*)d_in[1];   // [512]
  const float* w1 = (const float*)d_in[2];   // [512]

  _Float16* E = (_Float16*)d_ws;             // 16 MB normalized f16 embeddings

  int*   rowsOut = (int*)d_out;                              // N*31 int32
  int*   colsOut = rowsOut + (size_t)NN * KNN;               // N*31 int32
  float* valsOut = (float*)d_out + 2 * (size_t)NN * KNN;     // N*31 f32

  embed_normalize_kernel<<<NN, 256, 0, stream>>>(F, w0, w1, E);
  simtopk_kernel<<<NN / 32, 256, 0, stream>>>(E, rowsOut, colsOut, valsOut);
}
